// AttentionBlock_37357625540614
// MI455X (gfx1250) — matrix operand order repaired and checked by reading, not compiled, not run on hardware
//
#include <hip/hip_runtime.h>

// ---------------------------------------------------------------------------
// Attention block for MI455X (gfx1250, wave32, WMMA + TDM).
// fp32 -> bf16 convert, software-pipelined WMMA projection GEMMs (relu),
// fused flash-attention with online softmax. K-chunk staging uses the
// Tensor Data Mover (tensor_load_to_lds + s_wait_tensorcnt); next-chunk V is
// warmed with global_prefetch_b8.
// ---------------------------------------------------------------------------

typedef __attribute__((ext_vector_type(16))) __bf16 v16bf;
typedef __attribute__((ext_vector_type(8)))  float  v8f;
typedef __attribute__((ext_vector_type(4)))  unsigned v4u;
typedef __attribute__((ext_vector_type(8)))  int      v8i;
typedef __attribute__((ext_vector_type(4)))  int      v4i;

struct alignas(16) U4 { unsigned x, y, z, w; };          // POD 16-byte chunk
union  V16 { v16bf v; U4 q[2]; unsigned short s[16]; };  // 32B = one WMMA A/B operand
union  V8  { U4 q;    unsigned short s[8]; };

static constexpr int BB = 8, NN = 2048, DD = 512, EE = 512;

__device__ __forceinline__ unsigned short f2bf(float f) {
  unsigned u = __float_as_uint(f);
  u += 0x7FFFu + ((u >> 16) & 1u);      // round-to-nearest-even
  return (unsigned short)(u >> 16);
}

// ---------------- Kernel 1a: x fp32 -> bf16 --------------------------------
__global__ __launch_bounds__(256) void k_convert_x(const float* __restrict__ x,
                                                   unsigned short* __restrict__ xb) {
  const int i = blockIdx.x * 256 + threadIdx.x;     // each thread: 4 elements
  const float4 ld = reinterpret_cast<const float4*>(x)[i];
  V8 st;
  st.s[0] = f2bf(ld.x); st.s[1] = f2bf(ld.y);
  st.s[2] = f2bf(ld.z); st.s[3] = f2bf(ld.w);
  reinterpret_cast<unsigned long long*>(xb)[i] =
      *reinterpret_cast<unsigned long long*>(&st);
}

// ---------------- Kernel 1b: W -> W^T bf16 ---------------------------------
__global__ __launch_bounds__(256) void k_transpose_w(const float* __restrict__ wq,
                                                     const float* __restrict__ wk,
                                                     const float* __restrict__ wv,
                                                     unsigned short* __restrict__ wt) {
  const int z = blockIdx.y;
  const float* W = (z == 0) ? wq : ((z == 1) ? wk : wv);
  const int flat = blockIdx.x * 256 + threadIdx.x;  // 0 .. 512*512-1
  const int d = flat >> 9, e = flat & 511;
  wt[(size_t)z * DD * EE + (size_t)e * DD + d] = f2bf(W[flat]);
}

// ---------------- Kernel 2: QKV projection GEMM (relu) ---------------------
// grid (M/64, E/64, 3), block 128 (4 waves); wave w -> 16 rows x 64 cols.
// Software-pipelined: chunk c+1 operands are fetched before chunk c's WMMAs.
__global__ __launch_bounds__(128) void k_proj(const unsigned short* __restrict__ xb,
                                              const unsigned short* __restrict__ wt,
                                              const float* __restrict__ bq,
                                              const float* __restrict__ bk,
                                              const float* __restrict__ bv,
                                              unsigned short* __restrict__ oq,
                                              unsigned short* __restrict__ ok,
                                              unsigned short* __restrict__ ov) {
  const int z = blockIdx.z;
  const unsigned short* W = wt + (size_t)z * DD * EE;
  const float* bias       = (z == 0) ? bq : ((z == 1) ? bk : bv);
  unsigned short* outp    = (z == 0) ? oq : ((z == 1) ? ok : ov);

  const int tid = threadIdx.x, wav = tid >> 5, lane = tid & 31;
  const int hf = lane >> 4, ln = lane & 15;
  const int m0 = blockIdx.x * 64 + wav * 16;
  const int n0 = blockIdx.y * 64;

  v8f acc[4];
#pragma unroll
  for (int t = 0; t < 4; ++t)
#pragma unroll
    for (int r = 0; r < 8; ++r) acc[t][r] = 0.f;

  const U4* xrow = reinterpret_cast<const U4*>(xb + (size_t)(m0 + ln) * DD);
  const U4* wrowp[4];
#pragma unroll
  for (int nt = 0; nt < 4; ++nt)
    wrowp[nt] = reinterpret_cast<const U4*>(W + (size_t)(n0 + nt * 16 + ln) * DD);

  // prologue: chunk 0 operands
  V16 A, Bt[4];
  A.q[0] = xrow[hf * 2];
  A.q[1] = xrow[hf * 2 + 1];
#pragma unroll
  for (int nt = 0; nt < 4; ++nt) {
    Bt[nt].q[0] = wrowp[nt][hf * 2];
    Bt[nt].q[1] = wrowp[nt][hf * 2 + 1];
  }

#pragma unroll
  for (int c = 0; c < 16; ++c) {               // K chunks of 32
    V16 An, Bn[4];
    if (c < 15) {                              // prefetch next chunk's operands
      An.q[0] = xrow[(c + 1) * 4 + hf * 2];
      An.q[1] = xrow[(c + 1) * 4 + hf * 2 + 1];
#pragma unroll
      for (int nt = 0; nt < 4; ++nt) {
        Bn[nt].q[0] = wrowp[nt][(c + 1) * 4 + hf * 2];
        Bn[nt].q[1] = wrowp[nt][(c + 1) * 4 + hf * 2 + 1];
      }
    }
#pragma unroll
    for (int nt = 0; nt < 4; ++nt)
      acc[nt] = __builtin_amdgcn_wmma_f32_16x16x32_bf16(
          false, A.v, false, Bt[nt].v, (short)0, acc[nt], false, false);
    if (c < 15) {
      A = An;
#pragma unroll
      for (int nt = 0; nt < 4; ++nt) Bt[nt] = Bn[nt];
    }
  }

#pragma unroll
  for (int nt = 0; nt < 4; ++nt) {
    const int col = n0 + nt * 16 + ln;
    const float bb = bias[col];
#pragma unroll
    for (int r = 0; r < 8; ++r) {
      float vv = acc[nt][r] + bb;
      vv = vv > 0.f ? vv : 0.f;                // relu
      const int row = m0 + r + 8 * hf;         // C/D layout
      outp[(size_t)row * EE + col] = f2bf(vv);
    }
  }
}

// ---------------- Kernel 3: fused flash attention --------------------------
// grid (N/64, B), block 128 (4 waves); wave owns 16 q-rows x 512 v-cols.
// K chunk staged by the Tensor Data Mover; V chunk prefetched then transposed.
__global__ __launch_bounds__(128) void k_attn(const unsigned short* __restrict__ Qb,
                                              const unsigned short* __restrict__ Kb,
                                              const unsigned short* __restrict__ Vb,
                                              float* __restrict__ out) {
  __shared__ __align__(16) unsigned short KV[32 * 512];    // K chunk, then V^T chunk
  __shared__ __align__(16) unsigned short Psh[4][16 * 32]; // per-wave P relayout

  const int tid = threadIdx.x, wav = tid >> 5, lane = tid & 31;
  const int hf = lane >> 4, ln = lane & 15;
  const int b = blockIdx.y;
  const int q0 = blockIdx.x * 64 + wav * 16;
  const float SCALE = 0.044194173824159216f;   // 1/sqrt(512)

  // Q tile resident in registers, A-layout, 16 k-chunks of 32.
  V16 Qa[16];
  {
    const U4* qrow = reinterpret_cast<const U4*>(Qb + ((size_t)b * NN + q0 + ln) * EE);
#pragma unroll
    for (int c = 0; c < 16; ++c) {
      Qa[c].q[0] = qrow[c * 4 + hf * 2];
      Qa[c].q[1] = qrow[c * 4 + hf * 2 + 1];
    }
  }

  v8f acc[32];
#pragma unroll
  for (int t = 0; t < 32; ++t)
#pragma unroll
    for (int r = 0; r < 8; ++r) acc[t][r] = 0.f;

  float mrow[8], lrow[8];
#pragma unroll
  for (int r = 0; r < 8; ++r) { mrow[r] = -__builtin_inff(); lrow[r] = 0.f; }

  for (int kc = 0; kc < NN; kc += 32) {        // 64 key chunks
    __syncthreads();                           // prior PV reads of KV done

    // ---- stage K chunk (32 keys x 512 bf16, row-major) via TDM ----
    if (wav == 0) {
      const unsigned lds_base = (unsigned)(size_t)(&KV[0]);
      const unsigned long long ga =
          (unsigned long long)(size_t)(Kb + (((size_t)b * NN + kc) << 9));
      v4u g0;
      g0[0] = 1u;                                    // count=1 (valid), user mode
      g0[1] = lds_base;                              // lds_addr
      g0[2] = (unsigned)ga;                          // global_addr[31:0]
      g0[3] = (unsigned)(ga >> 32) | (2u << 30);     // global_addr[56:32] | type=2
      v8i g1;
      g1[0] = (int)(1u << 16);                       // data_size=2B, wg_mask=0
      g1[1] = (int)((unsigned)DD << 16);             // tensor_dim0 = 512
      g1[2] = (int)((unsigned)NN << 16);             // tensor_dim1 = 2048
      g1[3] = (int)((unsigned)DD << 16);             // tile_dim0 = 512
      g1[4] = 32;                                    // tile_dim1 = 32
      g1[5] = DD;                                    // tensor_dim0_stride = 512
      g1[6] = 0;                                     // dim1_stride[15:0] = 0
      g1[7] = (int)(((unsigned long long)NN * DD) >> 16);  // dim1_stride[47:16]
      v4i gz4 = {0, 0, 0, 0};
      v8i gz8 = {0, 0, 0, 0, 0, 0, 0, 0};
      __builtin_amdgcn_tensor_load_to_lds(g0, g1, gz4, gz4, gz8, 0);
      __builtin_amdgcn_s_wait_tensorcnt(0);
    }
    // warm L2 with next chunk's V while the TDM runs
    if (kc + 32 < NN) {
      const unsigned short* nv = Vb + (((size_t)b * NN + kc + 32) << 9);
      __builtin_prefetch(nv + tid * 128, 0, 1);
      __builtin_prefetch(nv + tid * 128 + 64, 0, 1);
    }
    __syncthreads();

    // ---- S = Q . K^T over 512 features ----
    v8f S0, S1;
#pragma unroll
    for (int r = 0; r < 8; ++r) { S0[r] = 0.f; S1[r] = 0.f; }
    const U4* kv4 = reinterpret_cast<const U4*>(KV);
#pragma unroll
    for (int c = 0; c < 16; ++c) {
      V16 B0, B1;
      B0.q[0] = kv4[ln * 64 + c * 4 + hf * 2];
      B0.q[1] = kv4[ln * 64 + c * 4 + hf * 2 + 1];
      B1.q[0] = kv4[(16 + ln) * 64 + c * 4 + hf * 2];
      B1.q[1] = kv4[(16 + ln) * 64 + c * 4 + hf * 2 + 1];
      S0 = __builtin_amdgcn_wmma_f32_16x16x32_bf16(false, Qa[c].v, false, B0.v,
                                                   (short)0, S0, false, false);
      S1 = __builtin_amdgcn_wmma_f32_16x16x32_bf16(false, Qa[c].v, false, B1.v,
                                                   (short)0, S1, false, false);
    }

    // ---- online softmax (rows live in VGPR index; cols in 16-lane halves) ----
#pragma unroll
    for (int r = 0; r < 8; ++r) {
      float t0 = S0[r] * SCALE, t1 = S1[r] * SCALE;
      float mx = fmaxf(t0, t1);
#pragma unroll
      for (int o = 8; o >= 1; o >>= 1) mx = fmaxf(mx, __shfl_xor(mx, o, 32));
      const float mnew = fmaxf(mrow[r], mx);
      const float p0 = __expf(t0 - mnew);
      const float p1 = __expf(t1 - mnew);
      float rs = p0 + p1;
#pragma unroll
      for (int o = 8; o >= 1; o >>= 1) rs += __shfl_xor(rs, o, 32);
      const float corr = __expf(mrow[r] - mnew);
      lrow[r] = lrow[r] * corr + rs;
      mrow[r] = mnew;
      const int prow = (r + 8 * hf) * 32;      // C-layout row -> P_sh row
      Psh[wav][prow + ln]      = f2bf(p0);
      Psh[wav][prow + 16 + ln] = f2bf(p1);
#pragma unroll
      for (int t = 0; t < 32; ++t) acc[t][r] *= corr;
    }

    __syncthreads();
    // ---- stage V^T chunk (512 cols x 32 keys) into same LDS buffer ----
#pragma unroll
    for (int i = 0; i < 16; ++i) {
      const int flat = (i * 128 + tid) * 8;
      const int row = flat >> 9, col = flat & 511;
      V8 tmp;
      tmp.q = *reinterpret_cast<const U4*>(Vb + (((size_t)b * NN + kc + row) << 9) + col);
#pragma unroll
      for (int j = 0; j < 8; ++j) KV[((col + j) << 5) + row] = tmp.s[j];
    }
    __syncthreads();

    // ---- reload P in A-layout from LDS ----
    V16 Pa;
    const U4* p4 = reinterpret_cast<const U4*>(Psh[wav]);
    Pa.q[0] = p4[ln * 4 + hf * 2];
    Pa.q[1] = p4[ln * 4 + hf * 2 + 1];

    // ---- O += P . V over this 32-key chunk, all 32 column tiles ----
    const U4* v4 = reinterpret_cast<const U4*>(KV);
#pragma unroll
    for (int t = 0; t < 32; ++t) {
      V16 Bv;
      Bv.q[0] = v4[(t * 16 + ln) * 4 + hf * 2];
      Bv.q[1] = v4[(t * 16 + ln) * 4 + hf * 2 + 1];
      acc[t] = __builtin_amdgcn_wmma_f32_16x16x32_bf16(false, Pa.v, false, Bv.v,
                                                       (short)0, acc[t], false, false);
    }
  }

  // ---- epilogue: O / l, fp32 out ----
#pragma unroll
  for (int r = 0; r < 8; ++r) {
    const float inv = 1.0f / lrow[r];
    const int row = q0 + r + 8 * hf;
    float* op = out + ((size_t)b * NN + row) * EE + ln;
#pragma unroll
    for (int t = 0; t < 32; ++t) op[t * 16] = acc[t][r] * inv;
  }
}

// ---------------------------------------------------------------------------
extern "C" void kernel_launch(void* const* d_in, const int* in_sizes, int n_in,
                              void* d_out, int out_size, void* d_ws, size_t ws_size,
                              hipStream_t stream) {
  const float* x  = (const float*)d_in[0];
  const float* Wq = (const float*)d_in[1];
  const float* bq = (const float*)d_in[2];
  const float* Wk = (const float*)d_in[3];
  const float* bk = (const float*)d_in[4];
  const float* Wv = (const float*)d_in[5];
  const float* bv = (const float*)d_in[6];
  float* out = (float*)d_out;

  char* ws = (char*)d_ws;
  const size_t OFF_XB = 0;
  const size_t OFF_WT = (size_t)BB * NN * DD * 2;          // x_bf16
  const size_t OFF_QB = OFF_WT + (size_t)3 * DD * EE * 2;  // 3 transposed weights
  const size_t OFF_KB = OFF_QB + (size_t)BB * NN * EE * 2;
  const size_t OFF_VB = OFF_KB + (size_t)BB * NN * EE * 2;

  unsigned short* xb = (unsigned short*)(ws + OFF_XB);
  unsigned short* wt = (unsigned short*)(ws + OFF_WT);
  unsigned short* qb = (unsigned short*)(ws + OFF_QB);
  unsigned short* kb = (unsigned short*)(ws + OFF_KB);
  unsigned short* vb = (unsigned short*)(ws + OFF_VB);

  k_convert_x<<<(BB * NN * DD / 4) / 256, 256, 0, stream>>>(x, xb);
  k_transpose_w<<<dim3(DD * EE / 256, 3), 256, 0, stream>>>(Wq, Wk, Wv, wt);
  k_proj<<<dim3(BB * NN / 64, EE / 64, 3), 128, 0, stream>>>(xb, wt, bq, bk, bv,
                                                             qb, kb, vb);
  k_attn<<<dim3(NN / 64, BB), 128, 0, stream>>>(qb, kb, vb, out);
}
